// MixtureStochasticConvBlock_34222299414833
// MI455X (gfx1250) — matrix operand order repaired and checked by reading, not compiled, not run on hardware
//
#include <hip/hip_runtime.h>
#include <cstdint>
#include <cstddef>

typedef __attribute__((ext_vector_type(16))) _Float16 v16h;
typedef __attribute__((ext_vector_type(8)))  float    v8f;

#define CONST_LOG2PI 1.8378770664093453f

static constexpr int BATCH = 256;

// d_out float offsets (return-order concatenation)
static constexpr size_t OUT_OFF = 0;                         // (256,256,16,16)
static constexpr size_t Z_OFF   = 16777216;                  // (256,128,16,16)
static constexpr size_t LPP_OFF = 25165824;                  // (256,128,16,16,4)
static constexpr size_t LPQ_OFF = 58720256;                  // (256,128,16,16)
static constexpr size_t KLT_OFF = 67108864;                  // scalar
static constexpr size_t QMU_OFF = 67108865;                  // (256,128,16,16)
static constexpr size_t QLV_OFF = 75497473;                  // (256,128,16,16)
static constexpr size_t Y_OFF   = 83886081;                  // (256,4)
static constexpr size_t CE_OFF  = 83887105;                  // scalar

// ---------------------------------------------------------------------------
// Weight packing into A fragments, this file's contiguous k order (16-bit A 16x32):
//   lane<16 : K = 0..15 (v0-7) ; lane>=16 : K = 16..31 (v0-7)
// Flat index: (((coB*9 + rs)*NCB + cb)*512) + lane*16 + e
// ---------------------------------------------------------------------------
template<int CIN, int COUT>
__global__ void pack_weights(const float* __restrict__ w, _Float16* __restrict__ wp) {
  constexpr int NCB = CIN / 32;
  int idx = blockIdx.x * 256 + threadIdx.x;
  if (idx >= COUT * CIN * 9) return;
  int e    = idx & 15;
  int lane = (idx >> 4) & 31;
  int blk  = idx >> 9;
  int cb   = blk % NCB;
  int q    = blk / NCB;
  int rs   = q % 9;
  int coB  = q / 9;
  int co   = coB * 16 + (lane & 15);
  int base = (lane < 16) ? 0 : 16;
  int kl   = base + e;
  int ci   = cb * 32 + kl;
  wp[idx] = (_Float16)w[((size_t)co * CIN + ci) * 9 + rs];
}

// 8 async b128 DMA ops: 128 B of this thread's pixel row -> LDS (ASYNCcnt)
__device__ __forceinline__ void stage_chunk(const _Float16* src, uint32_t dsta) {
  const unsigned long long ga = (unsigned long long)(uintptr_t)src;
  #pragma unroll
  for (int j = 0; j < 8; ++j)
    asm volatile("global_load_async_to_lds_b128 %0, %1, off offset:%2"
                 :: "v"(dsta), "v"(ga), "i"(j * 16) : "memory");
}

// ---------------------------------------------------------------------------
// Implicit-GEMM 3x3 conv (pad=1, H=W=16) via v_wmma_f32_16x16x32_f16.
// Input f16 pixel-major [B][256][CIN]; 64-channel chunks double-buffered in
// LDS (2 x 32KB ping-pong) with async global->LDS DMA overlapping compute.
// Branchless boundary handling: combined (yy,xx) bounds mask via cndmask,
// WMMA issued unconditionally (no exec-divergent accumulator copies).
// MODE 0: relu -> f16 [B][256][COUT] (LDS-repacked b128 stores)
// MODE 1: clamp(-10,10); co<COUT/2 -> oA (q_mu NCHW), else -> oB (q_lv NCHW)
// MODE 2: f32 NCHW -> oA
// ---------------------------------------------------------------------------
template<int CIN, int COUT, int MODE>
__launch_bounds__(256)
__global__ void conv3x3_wmma(const _Float16* __restrict__ in16,
                             const _Float16* __restrict__ wpack,
                             const float* __restrict__ bias,
                             _Float16* __restrict__ o16,
                             float* __restrict__ oA,
                             float* __restrict__ oB) {
  extern __shared__ _Float16 lds[];                 // 2 x 16384 halfs
  constexpr int NCB = CIN / 32;                     // 32-ch K-blocks total
  constexpr int NCH = CIN / 64;                     // 64-ch chunks
  const int b    = blockIdx.y;
  const int coB  = blockIdx.x;
  const int tid  = threadIdx.x;
  const int wave = tid >> 5;
  const int lane = tid & 31;
  const int col  = lane & 15;                       // N column within tile
  const int hiH  = lane >> 4;                       // K/M half selector
  const _Float16* inb = in16 + (size_t)b * 256 * CIN;
  const uint32_t dsbase = (uint32_t)(uintptr_t)lds + (uint32_t)tid * 128u;

  v8f acc[2] = {{}, {}};
  const int n0 = wave * 32;                         // this wave owns pixels n0..n0+31

  // prologue: chunk 0 -> buffer 0
  stage_chunk(inb + (size_t)tid * CIN, dsbase);
  asm volatile("s_wait_asynccnt 0x0" ::: "memory");
  __syncthreads();

  for (int ch = 0; ch < NCH; ++ch) {
    if (ch + 1 < NCH)                               // prefetch next chunk
      stage_chunk(inb + (size_t)tid * CIN + (ch + 1) * 64,
                  dsbase + (uint32_t)(((ch + 1) & 1) * 32768));
    const _Float16* cbuf = lds + (ch & 1) * 16384;

    #pragma unroll
    for (int rs = 0; rs < 9; ++rs) {
      const int dy = rs / 3 - 1;
      const int dx = rs % 3 - 1;
      #pragma unroll
      for (int kb = 0; kb < 2; ++kb) {
        const int cbG = ch * 2 + kb;
        const v16h a = *(const v16h*)(wpack +
            (size_t)(((coB * 9 + rs) * NCB + cbG) * 512 + lane * 16));
        #pragma unroll
        for (int t = 0; t < 2; ++t) {
          const int yy = wave * 2 + t + dy;
          const int xx = col + dx;
          v16h bf = *(const v16h*)(cbuf +
              ((yy * 16 + xx) * 64 + kb * 32 + hiH * 16));
          if (dy != 0 || dx != 0) {                 // branchless boundary mask
            const bool ok = ((unsigned)yy < 16u) & ((unsigned)xx < 16u);
            v16h zero = {};
            bf = ok ? bf : zero;
          }
          acc[t] = __builtin_amdgcn_wmma_f32_16x16x32_f16(
              false, a, false, bf, (short)0, acc[t], false, false);
        }
      }
    }
    // my next-chunk DMA done + everyone finished reading current buffer
    asm volatile("s_wait_asynccnt 0x0" ::: "memory");
    __syncthreads();
  }

  // epilogue (C/D layout: VGPR i -> M=i (lanes 0-15) or M=8+i (lanes 16-31))
  if (MODE == 0) {
    // repack through wave-private LDS so stores are contiguous b128
    _Float16* sw = lds + wave * 512;
    #pragma unroll
    for (int t = 0; t < 2; ++t) {
      #pragma unroll
      for (int i = 0; i < 8; ++i) {
        const int m  = hiH ? (8 + i) : i;
        float v = acc[t][i] + bias[coB * 16 + m];
        v = v > 0.f ? v : 0.f;
        sw[(t * 16 + col) * 16 + m] = (_Float16)v;
      }
    }
    const int p = n0 + lane;
    uint4 w0 = *(uint4*)(sw + lane * 16);
    uint4 w1 = *(uint4*)(sw + lane * 16 + 8);
    uint4* dp = (uint4*)(o16 + ((size_t)b * 256 + p) * COUT + coB * 16);
    dp[0] = w0;
    dp[1] = w1;
  } else {
    #pragma unroll
    for (int t = 0; t < 2; ++t) {
      const int p = n0 + t * 16 + col;
      #pragma unroll
      for (int i = 0; i < 8; ++i) {
        const int m  = hiH ? (8 + i) : i;
        const int co = coB * 16 + m;
        float v = acc[t][i] + bias[co];
        if (MODE == 1) {
          v = v < -10.f ? -10.f : (v > 10.f ? 10.f : v);
          if (co < COUT / 2) oA[((size_t)b * (COUT / 2) + co) * 256 + p] = v;
          else               oB[((size_t)b * (COUT / 2) + (co - COUT / 2)) * 256 + p] = v;
        } else {
          oA[((size_t)b * COUT + co) * 256 + p] = v;
        }
      }
    }
  }
}

// ---------------------------------------------------------------------------
// f32 NCHW -> f16 pixel-major, optional FiLM (gamma*x+beta); coalesced stores
// ---------------------------------------------------------------------------
template<int C, bool MOD>
__global__ void pixmaj_f16(const float* __restrict__ src, _Float16* __restrict__ dst,
                           const float* __restrict__ gamma_ws,
                           const float* __restrict__ beta_ws) {
  size_t idx = (size_t)blockIdx.x * 256 + threadIdx.x;   // B*256*(C/8) threads
  int cg = (int)(idx % (C / 8));
  size_t t = idx / (C / 8);
  int p = (int)(t & 255);
  int b = (int)(t >> 8);
  const float* s = src + ((size_t)b * C + (size_t)cg * 8) * 256 + p;
  _Float16 outv[8];
  #pragma unroll
  for (int j = 0; j < 8; ++j) {
    float v = s[(size_t)j * 256];
    if (MOD) {
      int c = cg * 8 + j;
      v = gamma_ws[b * C + c] * v + beta_ws[b * C + c];
    }
    outv[j] = (_Float16)v;
  }
  *(uint4*)(dst + ((size_t)b * 256 + p) * C + (size_t)cg * 8) = *(uint4*)outv;
}

// ---------------------------------------------------------------------------
// qy head: logits (4-way dot over 32768), gumbel softmax y, CE, JS, FiLM params
// ---------------------------------------------------------------------------
__launch_bounds__(256)
__global__ void qy_kernel(const _Float16* __restrict__ h16,      // [B][256][128]
                          const float* __restrict__ lin_w,       // [4][32768]
                          const float* __restrict__ lin_b,
                          const float* __restrict__ u_gumbel,    // [B][4]
                          const int*   __restrict__ label,
                          const float* __restrict__ gamma_w, const float* __restrict__ gamma_b,
                          const float* __restrict__ beta_w,  const float* __restrict__ beta_b,
                          float* __restrict__ gamma_ws,          // [B][256]
                          float* __restrict__ beta_ws,
                          float* __restrict__ y_out,
                          float* __restrict__ ce_accum,
                          float* __restrict__ js_accum) {
  __shared__ float red[256];
  __shared__ float logits_s[4];
  const int b = blockIdx.x, tid = threadIdx.x;
  const _Float16* hb = h16 + (size_t)b * 32768;

  float part[4] = {0.f, 0.f, 0.f, 0.f};
  for (int j = tid; j < 32768; j += 256) {           // coalesced h16 reads
    float hv = (float)hb[j];
    int co = j & 127, p = j >> 7;
    int i = co * 256 + p;                            // reference flat index
    #pragma unroll
    for (int k = 0; k < 4; ++k) part[k] += hv * lin_w[(size_t)k * 32768 + i];
  }
  #pragma unroll
  for (int k = 0; k < 4; ++k) {
    red[tid] = part[k];
    __syncthreads();
    for (int s = 128; s > 0; s >>= 1) {
      if (tid < s) red[tid] += red[tid + s];
      __syncthreads();
    }
    if (tid == 0) logits_s[k] = red[0] + lin_b[k];
    __syncthreads();
  }

  if (tid == 0) {
    float lg[4], yv[4];
    for (int k = 0; k < 4; ++k) lg[k] = logits_s[k];
    float pert[4], mx = -1e30f;
    for (int k = 0; k < 4; ++k) {
      float g = -logf(-logf(u_gumbel[b * 4 + k]));
      pert[k] = lg[k] + g;
      mx = pert[k] > mx ? pert[k] : mx;
    }
    float se = 0.f;
    for (int k = 0; k < 4; ++k) { yv[k] = expf(pert[k] - mx); se += yv[k]; }
    for (int k = 0; k < 4; ++k) { yv[k] /= se; y_out[b * 4 + k] = yv[k]; }
    float lmx = -1e30f;
    for (int k = 0; k < 4; ++k) lmx = lg[k] > lmx ? lg[k] : lmx;
    float ls = 0.f;
    for (int k = 0; k < 4; ++k) ls += expf(lg[k] - lmx);
    float lse = logf(ls) + lmx;
    atomicAdd(ce_accum, -(lg[label[b]] - lse));
    float js = 0.f;
    for (int k = 0; k < 4; ++k) {
      float m = 0.5f * (yv[k] + 0.25f);
      js += 0.5f * yv[k] * logf(yv[k] / (m + 1e-10f))
          + 0.5f * 0.25f * logf(0.25f / (m + 1e-10f));
    }
    atomicAdd(js_accum, js);
  }
  __syncthreads();

  {
    const int c = tid;
    float gs = gamma_b[c], bs = beta_b[c];
    #pragma unroll
    for (int k = 0; k < 4; ++k) {
      gs += logits_s[k] * gamma_w[c * 4 + k];
      bs += logits_s[k] * beta_w[c * 4 + k];
    }
    gamma_ws[b * 256 + c] = gs;
    beta_ws[b * 256 + c]  = bs;
  }
}

// ---------------------------------------------------------------------------
// z = mu + std*eps ; logprob_q ; z16 for out-conv
// ---------------------------------------------------------------------------
__global__ void z_kernel(const float* __restrict__ qmu,
                         const float* __restrict__ qlv,
                         const float* __restrict__ eps,
                         float* __restrict__ z_out,
                         float* __restrict__ lpq_out,
                         _Float16* __restrict__ z16) {
  size_t idx = (size_t)blockIdx.x * 256 + threadIdx.x;   // [B][128][256] flat
  int p = (int)(idx & 255);
  int c = (int)((idx >> 8) & 127);
  int b = (int)(idx >> 15);
  float mu = qmu[idx], lv = qlv[idx];
  float sd = (lv < 0.f) ? expf(0.5f * lv) : (1.f + lv);
  float z = mu + sd * eps[idx];
  z_out[idx] = z;
  z16[((size_t)b * 256 + p) * 128 + c] = (_Float16)z;
  float t = (z - mu) / sd;
  lpq_out[idx] = -0.5f * t * t - logf(sd) - 0.5f * CONST_LOG2PI;
}

// ---------------------------------------------------------------------------
// logprob_p (all 4 comps per thread, float4 store) + label-gathered KL reduce
// ---------------------------------------------------------------------------
__launch_bounds__(256)
__global__ void klp_kernel(const float* __restrict__ p_params,
                           const float* __restrict__ z,
                           const float* __restrict__ qmu,
                           const float* __restrict__ qlv,
                           const int*   __restrict__ label,
                           float* __restrict__ lpp_out,
                           float* __restrict__ kl_accum) {
  __shared__ float red[256];
  const int tid = threadIdx.x;
  size_t idx = (size_t)blockIdx.x * 256 + tid;           // [B][128][256] flat
  int p = (int)(idx & 255);
  int c = (int)((idx >> 8) & 127);
  int b = (int)(idx >> 15);
  float zv = z[idx], qm = qmu[idx], ql = qlv[idx];
  float qs = (ql < 0.f) ? expf(0.5f * ql) : (1.f + ql);
  const int lab = label[b];
  size_t pbase = (size_t)b * 1024 * 256 + (size_t)c * 256 + p;

  float lpp[4];
  float klsel = 0.f;
  #pragma unroll
  for (int k = 0; k < 4; ++k) {
    float pmu = p_params[pbase + (size_t)(k * 128) * 256];
    float plv = p_params[pbase + (size_t)(512 + k * 128) * 256];
    pmu = pmu < -10.f ? -10.f : (pmu > 10.f ? 10.f : pmu);
    plv = plv < -10.f ? -10.f : (plv > 10.f ? 10.f : plv);
    float ps = (plv < 0.f) ? expf(0.5f * plv) : (1.f + plv);
    float t = (zv - pmu) / ps;
    lpp[k] = -0.5f * t * t - logf(ps) - 0.5f * CONST_LOG2PI;
    if (k == lab) {
      float d = qm - pmu;
      klsel = logf(ps) - logf(qs) + (qs * qs + d * d) / (2.f * ps * ps) - 0.5f;
    }
  }
  *(float4*)(lpp_out + idx * 4) = make_float4(lpp[0], lpp[1], lpp[2], lpp[3]);

  red[tid] = klsel;
  __syncthreads();
  for (int s = 128; s > 0; s >>= 1) {
    if (tid < s) red[tid] += red[tid + s];
    __syncthreads();
  }
  if (tid == 0) atomicAdd(kl_accum, red[0]);
}

__global__ void init_scalars(float* s) {
  if (threadIdx.x < 3) s[threadIdx.x] = 0.f;
}

__global__ void finalize_kernel(const float* __restrict__ s,    // kl, js, ce
                                float* __restrict__ klt_out,
                                float* __restrict__ ce_out) {
  if (threadIdx.x == 0) {
    *klt_out = s[0] * (1.f / (32768.f * 256.f)) + s[1] * (1.f / 256.f);
    *ce_out  = s[2] * (1.f / 256.f);
  }
}

// ---------------------------------------------------------------------------
extern "C" void kernel_launch(void* const* d_in, const int* in_sizes, int n_in,
                              void* d_out, int out_size, void* d_ws, size_t ws_size,
                              hipStream_t stream) {
  (void)in_sizes; (void)n_in; (void)out_size; (void)ws_size;
  const float* p_params  = (const float*)d_in[0];
  const float* q_params  = (const float*)d_in[1];
  const int*   label     = (const int*)d_in[2];
  const float* eps       = (const float*)d_in[3];
  const float* u_gumbel  = (const float*)d_in[4];
  const float* qy_conv_w = (const float*)d_in[5];
  const float* qy_conv_b = (const float*)d_in[6];
  const float* qy_lin_w  = (const float*)d_in[7];
  const float* qy_lin_b  = (const float*)d_in[8];
  const float* gamma_w   = (const float*)d_in[9];
  const float* gamma_b   = (const float*)d_in[10];
  const float* beta_w    = (const float*)d_in[11];
  const float* beta_b    = (const float*)d_in[12];
  const float* qz1_w     = (const float*)d_in[13];
  const float* qz1_b     = (const float*)d_in[14];
  const float* qz2_w     = (const float*)d_in[15];
  const float* qz2_b     = (const float*)d_in[16];
  const float* out_w     = (const float*)d_in[17];
  const float* out_b     = (const float*)d_in[18];
  float* out = (float*)d_out;

  char* ws = (char*)d_ws;
  size_t off = 0;
  auto alloc = [&](size_t bytes) -> void* {
    void* p = ws + off;
    off += (bytes + 255) & ~(size_t)255;
    return p;
  };
  _Float16* wp_qy   = (_Float16*)alloc((size_t)128 * 256 * 9 * 2);
  _Float16* wp_qz1  = (_Float16*)alloc((size_t)256 * 256 * 9 * 2);
  _Float16* wp_qz2  = (_Float16*)alloc((size_t)256 * 256 * 9 * 2);
  _Float16* wp_out  = (_Float16*)alloc((size_t)256 * 128 * 9 * 2);
  _Float16* arenaA  = (_Float16*)alloc((size_t)BATCH * 256 * 256 * 2);  // qp16 then qmod16
  _Float16* t1_16   = (_Float16*)alloc((size_t)BATCH * 256 * 256 * 2);
  _Float16* arenaC  = (_Float16*)alloc((size_t)BATCH * 256 * 128 * 2);  // h16 then z16
  float*    gamma_ws = (float*)alloc((size_t)BATCH * 256 * 4);
  float*    beta_ws  = (float*)alloc((size_t)BATCH * 256 * 4);
  float*    scalars  = (float*)alloc(16 * 4);   // kl, js, ce

  init_scalars<<<1, 32, 0, stream>>>(scalars);
  pack_weights<256, 128><<<(128 * 256 * 9 + 255) / 256, 256, 0, stream>>>(qy_conv_w, wp_qy);
  pack_weights<256, 256><<<(256 * 256 * 9 + 255) / 256, 256, 0, stream>>>(qz1_w, wp_qz1);
  pack_weights<256, 256><<<(256 * 256 * 9 + 255) / 256, 256, 0, stream>>>(qz2_w, wp_qz2);
  pack_weights<128, 256><<<(256 * 128 * 9 + 255) / 256, 256, 0, stream>>>(out_w, wp_out);

  // h = relu(conv1(q_params))
  pixmaj_f16<256, false><<<(BATCH * 256 * 32) / 256, 256, 0, stream>>>(
      q_params, arenaA, nullptr, nullptr);
  conv3x3_wmma<256, 128, 0><<<dim3(8, BATCH), 256, 65536, stream>>>(
      arenaA, wp_qy, qy_conv_b, arenaC, nullptr, nullptr);

  // qy head: logits, y, CE, JS, gamma/beta
  qy_kernel<<<BATCH, 256, 0, stream>>>(arenaC, qy_lin_w, qy_lin_b, u_gumbel, label,
                                       gamma_w, gamma_b, beta_w, beta_b,
                                       gamma_ws, beta_ws, out + Y_OFF,
                                       scalars + 2, scalars + 1);

  // q_mod = gamma*q + beta ; qz1 ; qz2 -> clamped q_mu/q_lv
  pixmaj_f16<256, true><<<(BATCH * 256 * 32) / 256, 256, 0, stream>>>(
      q_params, arenaA, gamma_ws, beta_ws);
  conv3x3_wmma<256, 256, 0><<<dim3(16, BATCH), 256, 65536, stream>>>(
      arenaA, wp_qz1, qz1_b, t1_16, nullptr, nullptr);
  conv3x3_wmma<256, 256, 1><<<dim3(16, BATCH), 256, 65536, stream>>>(
      t1_16, wp_qz2, qz2_b, nullptr, out + QMU_OFF, out + QLV_OFF);

  // z, logprob_q, z16
  z_kernel<<<(BATCH * 128 * 256) / 256, 256, 0, stream>>>(
      out + QMU_OFF, out + QLV_OFF, eps, out + Z_OFF, out + LPQ_OFF, arenaC);

  // out conv
  conv3x3_wmma<128, 256, 2><<<dim3(16, BATCH), 256, 65536, stream>>>(
      arenaC, wp_out, out_b, nullptr, out + OUT_OFF, nullptr);

  // logprob_p + KL(label) reduction
  klp_kernel<<<(BATCH * 128 * 256) / 256, 256, 0, stream>>>(
      p_params, out + Z_OFF, out + QMU_OFF, out + QLV_OFF, label,
      out + LPP_OFF, scalars + 0);

  finalize_kernel<<<1, 32, 0, stream>>>(scalars, out + KLT_OFF, out + CE_OFF);
}
